// TCN_GNN_82291573391513
// MI455X (gfx1250) — matrix operand order repaired and checked by reading, not compiled, not run on hardware
//
#include <hip/hip_runtime.h>
#include <hip/hip_bf16.h>
#include <math.h>

// Problem constants (match reference)
#define B_   2
#define T_   128
#define N_   512
#define CIN_ 16
#define C_   64
#define L_   4
#define KW_  3
#define E_   2048
#define ROWS (B_*T_*N_)          // 131072 rows of C_=64 features
#define EPS_ 1e-5f

typedef __bf16 bf16_t;
typedef __attribute__((ext_vector_type(16))) __bf16 v16bf;
typedef __attribute__((ext_vector_type(8)))  __bf16 v8bf;   // 16B, align 16
typedef __attribute__((ext_vector_type(8)))  float  v8f;
typedef __attribute__((ext_vector_type(4)))  float  v4f;    // 16B, align 16

// ---------------------------------------------------------------- WMMA helpers
__device__ inline v8f wmma_bf16(v16bf a, v16bf b, v8f c) {
  // D = A(16x32) x B(32x16) + C, f32 accumulate
  return __builtin_amdgcn_wmma_f32_16x16x32_bf16(false, a, false, b,
                                                 (short)0, c, false, false);
}

// A-fragment (16x32 bf16) loader. row_base = &A[m][0] for this lane's row m,
// must be 16B aligned. K mapping per ISA:
//   hi=0 : i<8 -> K=i,      i>=8 -> K=16+(i-8)
//   hi=1 : i<8 -> K=8+i,    i>=8 -> K=24+(i-8)
__device__ inline v16bf load_a(const bf16_t* row_base, int hi) {
  v8bf lo = *(const v8bf*)(row_base + hi * 8);        // 16B vector load
  v8bf up = *(const v8bf*)(row_base + 16 + hi * 8);   // 16B vector load
  v16bf a;
#pragma unroll
  for (int i = 0; i < 8; ++i) { a[i] = lo[i]; a[8 + i] = up[i]; }
  return a;
}

__device__ inline v16bf load_a_f32(const float* row_base, int hi) {
  const v4f* p0 = (const v4f*)(row_base + hi * 8);
  const v4f* p1 = (const v4f*)(row_base + 16 + hi * 8);
  v4f q0 = p0[0], q1 = p0[1], q2 = p1[0], q3 = p1[1];
  v16bf a;
#pragma unroll
  for (int i = 0; i < 4; ++i) {
    a[i]      = (bf16_t)q0[i];
    a[4 + i]  = (bf16_t)q1[i];
    a[8 + i]  = (bf16_t)q2[i];
    a[12 + i] = (bf16_t)q3[i];
  }
  return a;
}

__device__ inline v16bf zero_a() {
  v16bf a;
#pragma unroll
  for (int i = 0; i < 16; ++i) a[i] = (bf16_t)0.0f;
  return a;
}

// B-fragment (32x16 bf16) loader from weights packed transposed: Wt[n][k],
// row length kstride (multiple of 32). Column n = nbase + (lane&15),
// K = kbase + hi*8 + i (i<8), kbase + 16 + hi*8 + (i-8) (i>=8) -> two 16B vector loads.
__device__ inline v16bf load_b(const bf16_t* wt, int nbase, int kstride,
                               int kbase, int lane) {
  const bf16_t* p = wt + (size_t)(nbase + (lane & 15)) * kstride + kbase
                       + ((lane >> 4) & 1) * 8;
  v8bf lo = *(const v8bf*)(p);
  v8bf up = *(const v8bf*)(p + 16);
  v16bf b;
#pragma unroll
  for (int i = 0; i < 8; ++i) { b[i] = lo[i]; b[8 + i] = up[i]; }
  return b;
}

__device__ inline float sigmoidf_(float x) { return 1.0f / (1.0f + __expf(-x)); }
__device__ inline float siluf_(float x)    { return x * sigmoidf_(x); }

// ---------------------------------------------------------------- small utils
__global__ void zero_f32_kernel(float* p, int count) {
  int i = blockIdx.x * blockDim.x + threadIdx.x;
  if (i < count) p[i] = 0.0f;
}

// pack W(fi,fo) f32 row-major -> bf16 Wt[n][k]  (n=fo dim, k=fi dim)
__global__ void pack_wt_kernel(const float* __restrict__ W, bf16_t* __restrict__ out,
                               int fi, int fo) {
  int id = blockIdx.x * blockDim.x + threadIdx.x;
  if (id >= fi * fo) return;
  int k = id % fi, n = id / fi;
  out[(size_t)n * fi + k] = (bf16_t)W[(size_t)k * fo + n];
}

// pack conv_W(C,C,K)=[co][ci][kt] -> bf16 [kt][co][ci]
__global__ void pack_conv_kernel(const float* __restrict__ W, bf16_t* __restrict__ out) {
  int id = blockIdx.x * blockDim.x + threadIdx.x;
  if (id >= KW_ * C_ * C_) return;
  int ci = id % C_, co = (id / C_) % C_, kt = id / (C_ * C_);
  out[((size_t)kt * C_ + co) * C_ + ci] = (bf16_t)W[((size_t)co * C_ + ci) * KW_ + kt];
}

// ---------------------------------------------------------------- encoder
// h0 = x @ enc_W + enc_b + emb  (layout g[b][t][n][c])
__global__ void encoder_kernel(const float* __restrict__ x,
                               const float* __restrict__ enc_W,
                               const float* __restrict__ enc_b,
                               const float* __restrict__ emb,
                               float* __restrict__ g, bf16_t* __restrict__ gbf) {
  size_t i = (size_t)blockIdx.x * blockDim.x + threadIdx.x;
  if (i >= (size_t)ROWS * C_) return;
  int c = (int)(i & (C_ - 1));
  size_t row = i >> 6;
  int n = (int)(row % N_);
  float acc = enc_b[c] + emb[(size_t)n * C_ + c];
  const float* xr = x + row * CIN_;
#pragma unroll
  for (int k = 0; k < CIN_; ++k) acc += xr[k] * enc_W[k * C_ + c];
  g[i] = acc;
  gbf[i] = (bf16_t)acc;
}

// ---------------------------------------------------------------- dilated conv (WMMA)
// y[b,t,n,co] = sum_{ci,kt} W[co,ci,kt] * h[b, t-(2-kt)*dil, n, ci] + conv_b[co]
// One wave computes a 16(t) x 64(co) strip; 3 taps x 2 k-steps x 4 n-tiles = 24 WMMA.
__global__ void conv_wmma_kernel(const bf16_t* __restrict__ gbf,
                                 const bf16_t* __restrict__ wpack, // [3][64][64]
                                 const float* __restrict__ conv_b,
                                 float* __restrict__ y,
                                 float* __restrict__ ssum, float* __restrict__ ssq,
                                 int dil) {
  int wave = (blockIdx.x * blockDim.x + threadIdx.x) >> 5;   // 0..8191
  int lane = threadIdx.x & 31;
  int tt = wave & (T_ / 16 - 1);
  int bn = wave >> 3;
  int b = bn >> 9;              // N_=512
  int n = bn & (N_ - 1);
  int t0 = tt * 16;
  int m = lane & 15, hi = (lane >> 4) & 1;

  v8f acc[4];
#pragma unroll
  for (int nt = 0; nt < 4; ++nt)
#pragma unroll
    for (int r = 0; r < 8; ++r) acc[nt][r] = 0.0f;

#pragma unroll
  for (int kt = 0; kt < KW_; ++kt) {
    int tm = t0 + m - (2 - kt) * dil;
    bool valid = (tm >= 0);
    const bf16_t* abase =
        gbf + (((size_t)(b * T_ + (valid ? tm : 0)) * N_ + n) << 6);
#pragma unroll
    for (int ks = 0; ks < 2; ++ks) {
      v16bf a = valid ? load_a(abase + ks * 32, hi) : zero_a();
#pragma unroll
      for (int nt = 0; nt < 4; ++nt) {
        v16bf bb = load_b(wpack + (size_t)kt * C_ * C_, nt * 16, C_, ks * 32, lane);
        acc[nt] = wmma_bf16(a, bb, acc[nt]);
      }
    }
  }

#pragma unroll
  for (int nt = 0; nt < 4; ++nt) {
    int co = nt * 16 + (lane & 15);
    float bias = conv_b[co];
    float ps = 0.0f, pq = 0.0f;
#pragma unroll
    for (int r = 0; r < 8; ++r) {
      int trow = t0 + r + hi * 8;
      float v = acc[nt][r] + bias;
      y[(((size_t)(b * T_ + trow) * N_ + n) << 6) + co] = v;
      ps += v; pq += v * v;
    }
    atomicAdd(&ssum[co], ps);
    atomicAdd(&ssq[co], pq);
  }
}

// ---------------------------------------------------------------- BN + ReLU + residual
// Also zeroes the y buffer in place (it becomes the edge-scatter agg buffer).
__global__ void bn_relu_add_kernel(float* y_then_agg,
                                   const float* __restrict__ ssum,
                                   const float* __restrict__ ssq,
                                   const float* __restrict__ bn_g,
                                   const float* __restrict__ bn_b,
                                   float* __restrict__ skips,
                                   float* __restrict__ g, bf16_t* __restrict__ gbf) {
  size_t i = (size_t)blockIdx.x * blockDim.x + threadIdx.x;
  if (i >= (size_t)ROWS * C_) return;
  int c = (int)(i & (C_ - 1));
  const float cnt = (float)ROWS;
  float mu = ssum[c] / cnt;
  float var = ssq[c] / cnt - mu * mu;
  float yv = y_then_agg[i];
  y_then_agg[i] = 0.0f;                 // becomes zeroed agg for the GNN stage
  float v = (yv - mu) * rsqrtf(var + EPS_) * bn_g[c] + bn_b[c];
  v = fmaxf(v, 0.0f);
  skips[i] += v;
  float h = g[i] + v;
  g[i] = h;
  gbf[i] = (bf16_t)h;
}

// ---------------------------------------------------------------- edge MLP + scatter (WMMA)
// Per wave: 16 edges of one (b,t). cat(xi,xj)(128) @ mW1 -> silu -> @mW2 -> silu
// -> gate -> atomic scatter-add into agg at dst.
__global__ __launch_bounds__(128)
void edge_mlp_kernel(const bf16_t* __restrict__ gbf,
                     const int* __restrict__ edge_index,
                     const bf16_t* __restrict__ m1p,   // [32][128]
                     const float* __restrict__ mb1,
                     const bf16_t* __restrict__ m2p,   // [64][32]
                     const float* __restrict__ mb2,
                     const float* __restrict__ gWv, const float* __restrict__ gbv,
                     float* __restrict__ agg) {
  __shared__ __align__(16) bf16_t s_m1[4][16 * 32];
  __shared__ __align__(16) float  s_m2[4][16 * 64];
  __shared__ float  s_gate[4][16];
  __shared__ int    s_dst[4][16];

  int wave = (blockIdx.x * blockDim.x + threadIdx.x) >> 5;  // 0..32767
  int wv = (threadIdx.x >> 5) & 3;
  int lane = threadIdx.x & 31;
  int et = wave & (E_ / 16 - 1);
  int bt = wave >> 7;                 // (b*T+t)
  int e0 = et * 16;
  int m = lane & 15, hi = (lane >> 4) & 1;

  const int* srcI = edge_index;
  const int* dstI = edge_index + E_;
  int dstm = dstI[e0 + m];
  int srcm = srcI[e0 + m];
  if (lane < 16) s_dst[wv][lane] = dstm;

  const bf16_t* gbt = gbf + (((size_t)bt * N_) << 6);

  // ---- stage 1: (16x128) @ (128x32)
  v8f a1[2];
#pragma unroll
  for (int nt = 0; nt < 2; ++nt)
#pragma unroll
    for (int r = 0; r < 8; ++r) a1[nt][r] = 0.0f;

#pragma unroll
  for (int ks = 0; ks < 4; ++ks) {
    int idx = (ks < 2) ? dstm : srcm;
    int coff = (ks & 1) * 32;
    v16bf a = load_a(gbt + ((size_t)idx << 6) + coff, hi);
#pragma unroll
    for (int nt = 0; nt < 2; ++nt) {
      v16bf bb = load_b(m1p, nt * 16, 128, ks * 32, lane);
      a1[nt] = wmma_bf16(a, bb, a1[nt]);
    }
  }
#pragma unroll
  for (int nt = 0; nt < 2; ++nt) {
    int nn = nt * 16 + (lane & 15);
    float bias = mb1[nn];
#pragma unroll
    for (int r = 0; r < 8; ++r) {
      float v = siluf_(a1[nt][r] + bias);
      s_m1[wv][(r + hi * 8) * 32 + nn] = (bf16_t)v;
    }
  }

  // ---- stage 2: (16x32) @ (32x64)  (A fragment from LDS; wave-local DS is in-order)
  v8f a2[4];
#pragma unroll
  for (int nt = 0; nt < 4; ++nt)
#pragma unroll
    for (int r = 0; r < 8; ++r) a2[nt][r] = 0.0f;
  {
    v16bf af = load_a(&s_m1[wv][m * 32], hi);
#pragma unroll
    for (int nt = 0; nt < 4; ++nt) {
      v16bf bb = load_b(m2p, nt * 16, 32, 0, lane);
      a2[nt] = wmma_bf16(af, bb, a2[nt]);
    }
  }
#pragma unroll
  for (int nt = 0; nt < 4; ++nt) {
    int nn = nt * 16 + (lane & 15);
    float bias = mb2[nn];
#pragma unroll
    for (int r = 0; r < 8; ++r) {
      float v = siluf_(a2[nt][r] + bias);
      s_m2[wv][(r + hi * 8) * 64 + nn] = v;
    }
  }

  // ---- gate: sigmoid(m2 . gW + gb)
  if (lane < 16) {
    float d = gbv[0];
#pragma unroll 8
    for (int c = 0; c < 64; ++c) d += s_m2[wv][lane * 64 + c] * gWv[c];
    s_gate[wv][lane] = sigmoidf_(d);
  }

  // ---- scatter-add gate*m into agg[bt, dst, :]
  {
    int row = lane & 15;
    int d = s_dst[wv][row];
    float gt = s_gate[wv][row];
    float* ab = agg + (((size_t)bt * N_ + d) << 6);
#pragma unroll
    for (int j = 0; j < 32; ++j) {
      int c = hi * 32 + j;
      atomicAdd(&ab[c], gt * s_m2[wv][row * 64 + c]);
    }
  }
}

// ---------------------------------------------------------------- update MLP (WMMA)
// upd = silu(cat(agg,g) @ uW1 + ub1) @ uW2 + ub2 ; g = upd + g ; accumulate norm stats
__global__ __launch_bounds__(128)
void update_mlp_kernel(const float* __restrict__ agg,
                       const bf16_t* __restrict__ gbf,
                       const bf16_t* __restrict__ u1p,  // [64][128]
                       const float* __restrict__ ub1,
                       const bf16_t* __restrict__ u2p,  // [64][64]
                       const float* __restrict__ ub2,
                       float* __restrict__ g,
                       float* __restrict__ ssum, float* __restrict__ ssq) {
  __shared__ __align__(16) bf16_t s_u[4][16 * 64];

  int wave = (blockIdx.x * blockDim.x + threadIdx.x) >> 5;  // 0..8191
  int wv = (threadIdx.x >> 5) & 3;
  int lane = threadIdx.x & 31;
  int m = lane & 15, hi = (lane >> 4) & 1;
  size_t r0 = (size_t)wave * 16;
  size_t rowm = r0 + m;

  const float* arow = agg + (rowm << 6);
  const bf16_t* grow = gbf + (rowm << 6);

  v8f a1[4];
#pragma unroll
  for (int nt = 0; nt < 4; ++nt)
#pragma unroll
    for (int r = 0; r < 8; ++r) a1[nt][r] = 0.0f;

#pragma unroll
  for (int ks = 0; ks < 4; ++ks) {
    v16bf a = (ks < 2) ? load_a_f32(arow + ks * 32, hi)
                       : load_a(grow + (ks - 2) * 32, hi);
#pragma unroll
    for (int nt = 0; nt < 4; ++nt) {
      v16bf bb = load_b(u1p, nt * 16, 128, ks * 32, lane);
      a1[nt] = wmma_bf16(a, bb, a1[nt]);
    }
  }
#pragma unroll
  for (int nt = 0; nt < 4; ++nt) {
    int nn = nt * 16 + (lane & 15);
    float bias = ub1[nn];
#pragma unroll
    for (int r = 0; r < 8; ++r) {
      float v = siluf_(a1[nt][r] + bias);
      s_u[wv][(r + hi * 8) * 64 + nn] = (bf16_t)v;
    }
  }

  v8f a2[4];
#pragma unroll
  for (int nt = 0; nt < 4; ++nt)
#pragma unroll
    for (int r = 0; r < 8; ++r) a2[nt][r] = 0.0f;
#pragma unroll
  for (int ks = 0; ks < 2; ++ks) {
    v16bf af = load_a(&s_u[wv][m * 64 + ks * 32], hi);
#pragma unroll
    for (int nt = 0; nt < 4; ++nt) {
      v16bf bb = load_b(u2p, nt * 16, 64, ks * 32, lane);
      a2[nt] = wmma_bf16(af, bb, a2[nt]);
    }
  }

#pragma unroll
  for (int nt = 0; nt < 4; ++nt) {
    int c = nt * 16 + (lane & 15);
    float bias = ub2[c];
    float ps = 0.0f, pq = 0.0f;
#pragma unroll
    for (int r = 0; r < 8; ++r) {
      size_t row = r0 + r + hi * 8;
      float* gp = &g[(row << 6) + c];
      float v = a2[nt][r] + bias + *gp;
      *gp = v;
      ps += v; pq += v * v;
    }
    atomicAdd(&ssum[c], ps);
    atomicAdd(&ssq[c], pq);
  }
}

// ---------------------------------------------------------------- per-channel norm
__global__ void norm_kernel(const float* __restrict__ ssum, const float* __restrict__ ssq,
                            const float* __restrict__ n_g, const float* __restrict__ n_b,
                            float* __restrict__ g, bf16_t* __restrict__ gbf) {
  size_t i = (size_t)blockIdx.x * blockDim.x + threadIdx.x;
  if (i >= (size_t)ROWS * C_) return;
  int c = (int)(i & (C_ - 1));
  const float cnt = (float)ROWS;
  float mu = ssum[c] / cnt;
  float var = ssq[c] / cnt - mu * mu;
  float v = (g[i] - mu) * rsqrtf(var + EPS_) * n_g[c] + n_b[c];
  g[i] = v;
  gbf[i] = (bf16_t)v;
}

// ---------------------------------------------------------------- output heads
__global__ void head_kernel(const float* __restrict__ skips, const float* __restrict__ g,
                            const float* __restrict__ locW, const float* __restrict__ locb,
                            const float* __restrict__ scW, const float* __restrict__ scb,
                            float* __restrict__ out) {
  size_t row = (size_t)blockIdx.x * blockDim.x + threadIdx.x;
  if (row >= (size_t)ROWS) return;
  const float* sp = skips + (row << 6);
  const float* gp = g + (row << 6);
  float loc = locb[0], sc = scb[0];
#pragma unroll 8
  for (int c = 0; c < C_; ++c) {
    float r = sp[c] + gp[c];
    loc += r * locW[c];
    sc += r * scW[c];
  }
  float spv = (sc > 20.0f) ? sc : log1pf(__expf(sc));
  out[row * 2 + 0] = loc;
  out[row * 2 + 1] = spv;
}

// ---------------------------------------------------------------- host side
static inline int cdiv(long a, long b) { return (int)((a + b - 1) / b); }

extern "C" void kernel_launch(void* const* d_in, const int* in_sizes, int n_in,
                              void* d_out, int out_size, void* d_ws, size_t ws_size,
                              hipStream_t stream) {
  // Input flattening (setup_inputs dict, insertion order):
  // 0:x  1:enc_W 2:enc_b 3:emb
  // per layer l (base 4+16*l): conv_W, conv_b, bn_g, bn_b, mW1, mb1, mW2, mb2,
  //                            gW, gb, uW1, ub1, uW2, ub2, n_g, n_b
  // 68:locW 69:locb 70:scW 71:scb 72:edge_index
  const float* x     = (const float*)d_in[0];
  const float* enc_W = (const float*)d_in[1];
  const float* enc_b = (const float*)d_in[2];
  const float* emb   = (const float*)d_in[3];
  const float* locW  = (const float*)d_in[68];
  const float* locb  = (const float*)d_in[69];
  const float* scW   = (const float*)d_in[70];
  const float* scb   = (const float*)d_in[71];
  const int*   eidx  = (const int*)d_in[72];
  float* out = (float*)d_out;

  // workspace layout
  const size_t GSZ = (size_t)ROWS * C_ * sizeof(float);   // 32 MiB
  char* ws = (char*)d_ws;
  float*  g     = (float*)(ws);
  float*  skips = (float*)(ws + GSZ);
  float*  yagg  = (float*)(ws + 2 * GSZ);                 // shared: conv y / agg
  bf16_t* gbf   = (bf16_t*)(ws + 3 * GSZ);
  bf16_t* pk    = (bf16_t*)(ws + 3 * GSZ + GSZ / 2);
  float*  ssum  = (float*)(ws + 3 * GSZ + GSZ / 2 + 4 * 31744 * sizeof(bf16_t));
  float*  ssq   = ssum + C_;

  // packed weights per layer: conv[3*64*64], m1[32*128], m2[64*32], u1[64*128], u2[64*64]
  const size_t PKL = 3 * 64 * 64 + 32 * 128 + 64 * 32 + 64 * 128 + 64 * 64; // 30720
  for (int l = 0; l < L_; ++l) {
    int base = 4 + 16 * l;
    bf16_t* p = pk + (size_t)l * PKL;
    bf16_t* convp = p;
    bf16_t* m1p = convp + 3 * 64 * 64;
    bf16_t* m2p = m1p + 32 * 128;
    bf16_t* u1p = m2p + 64 * 32;
    bf16_t* u2p = u1p + 64 * 128;
    pack_conv_kernel<<<cdiv(3 * 64 * 64, 256), 256, 0, stream>>>(
        (const float*)d_in[base + 0], convp);
    pack_wt_kernel<<<cdiv(128 * 32, 256), 256, 0, stream>>>(
        (const float*)d_in[base + 4], m1p, 128, 32);
    pack_wt_kernel<<<cdiv(32 * 64, 256), 256, 0, stream>>>(
        (const float*)d_in[base + 6], m2p, 32, 64);
    pack_wt_kernel<<<cdiv(128 * 64, 256), 256, 0, stream>>>(
        (const float*)d_in[base + 10], u1p, 128, 64);
    pack_wt_kernel<<<cdiv(64 * 64, 256), 256, 0, stream>>>(
        (const float*)d_in[base + 12], u2p, 64, 64);
  }

  const long NG = (long)ROWS * C_;
  zero_f32_kernel<<<cdiv(NG, 256), 256, 0, stream>>>(skips, (int)NG);
  zero_f32_kernel<<<cdiv(NG, 256), 256, 0, stream>>>(yagg, (int)NG); // agg for layer 0
  encoder_kernel<<<cdiv(NG, 256), 256, 0, stream>>>(x, enc_W, enc_b, emb, g, gbf);

  for (int l = 0; l < L_; ++l) {
    int base = 4 + 16 * l;
    bf16_t* p = pk + (size_t)l * PKL;
    bf16_t* convp = p;
    bf16_t* m1p = convp + 3 * 64 * 64;
    bf16_t* m2p = m1p + 32 * 128;
    bf16_t* u1p = m2p + 64 * 32;
    bf16_t* u2p = u1p + 64 * 128;
    const float* conv_b = (const float*)d_in[base + 1];
    const float* bn_g = (const float*)d_in[base + 2];
    const float* bn_b = (const float*)d_in[base + 3];
    const float* mb1 = (const float*)d_in[base + 5];
    const float* mb2 = (const float*)d_in[base + 7];
    const float* gW = (const float*)d_in[base + 8];
    const float* gb = (const float*)d_in[base + 9];
    const float* ub1 = (const float*)d_in[base + 11];
    const float* ub2 = (const float*)d_in[base + 13];
    const float* n_g = (const float*)d_in[base + 14];
    const float* n_b = (const float*)d_in[base + 15];
    int dil = 1 << l;

    // --- TCN block (conv writes y into yagg; bn_relu_add consumes it and
    //     leaves yagg zeroed, ready to serve as agg for the GNN stage)
    zero_f32_kernel<<<1, 128, 0, stream>>>(ssum, 128);
    conv_wmma_kernel<<<8192 * 32 / 256, 256, 0, stream>>>(gbf, convp, conv_b,
                                                          yagg, ssum, ssq, dil);
    bn_relu_add_kernel<<<cdiv(NG, 256), 256, 0, stream>>>(yagg, ssum, ssq, bn_g,
                                                          bn_b, skips, g, gbf);

    // --- GNN block
    edge_mlp_kernel<<<32768 * 32 / 128, 128, 0, stream>>>(gbf, eidx, m1p, mb1,
                                                          m2p, mb2, gW, gb, yagg);
    zero_f32_kernel<<<1, 128, 0, stream>>>(ssum, 128);
    update_mlp_kernel<<<8192 * 32 / 128, 128, 0, stream>>>(yagg, gbf, u1p, ub1,
                                                           u2p, ub2, g, ssum, ssq);
    norm_kernel<<<cdiv(NG, 256), 256, 0, stream>>>(ssum, ssq, n_g, n_b, g, gbf);
  }

  head_kernel<<<cdiv(ROWS, 256), 256, 0, stream>>>(skips, g, locW, locb, scW,
                                                   scb, out);
  (void)in_sizes; (void)n_in; (void)out_size; (void)ws_size;
}
